// WorldModel_33397665694521
// MI455X (gfx1250) — matrix operand order repaired and checked by reading, not compiled, not run on hardware
//
#include <hip/hip_runtime.h>
#include <hip/hip_bf16.h>
#include <math.h>
#include <stdint.h>

// Model dims
#define B_    64
#define S_    256
#define OBS_  256
#define H_    1024
#define SM1   255
#define STOCH_ 1024
#define MPAD  16384   // 16320 rows padded to 128-row blocking

typedef __attribute__((ext_vector_type(16))) __bf16 v16bf;
typedef __attribute__((ext_vector_type(8)))  float  v8f;
typedef __attribute__((ext_vector_type(4)))  int    v4i_t;

// ---------------------------------------------------------------------------
// Data-carrying scheduling fences: tie fragment registers through an empty
// asm so (a) all fragments stay simultaneously live (no RA coalescing) and
// (b) consumers (WMMA) cannot hoist above the fence -> loads batch, one
// s_wait_dscnt drain per burst instead of one per WMMA.
// ---------------------------------------------------------------------------
__device__ __forceinline__ void pin2(v16bf& a, v16bf& b) {
  asm volatile("" : "+v"(a), "+v"(b));
}
__device__ __forceinline__ void pin6(v16bf& a, v16bf& b, v16bf& c,
                                     v16bf& d, v16bf& e, v16bf& f) {
  asm volatile("" : "+v"(a), "+v"(b), "+v"(c), "+v"(d), "+v"(e), "+v"(f));
}
__device__ __forceinline__ void pin8(v16bf& a, v16bf& b, v16bf& c, v16bf& d,
                                     v16bf& e, v16bf& f, v16bf& g, v16bf& h) {
  asm volatile("" : "+v"(a), "+v"(b), "+v"(c), "+v"(d),
                    "+v"(e), "+v"(f), "+v"(g), "+v"(h));
}

// ---------------------------------------------------------------------------
// CDNA5 async global->LDS copy (ASYNCcnt path), with safe fallback.
// Probe-verified signature: (int4 AS1*, int4 AS3*, int offset, int cpol).
// ---------------------------------------------------------------------------
#if __has_builtin(__builtin_amdgcn_global_load_async_to_lds_b128)
#define ASYNC_LDS 1
#else
#define ASYNC_LDS 0
#endif

typedef __attribute__((address_space(1))) v4i_t* as1_v4i_p;
typedef __attribute__((address_space(3))) v4i_t* as3_v4i_p;

__device__ __forceinline__ void async_cp16(const void* g, void* l) {
#if ASYNC_LDS
  __builtin_amdgcn_global_load_async_to_lds_b128(
      (as1_v4i_p)(uintptr_t)g,
      (as3_v4i_p)(unsigned)(uintptr_t)l,
      0, 0);
#else
  *(float4*)l = *(const float4*)g;
#endif
}
__device__ __forceinline__ void wait_async_copies() {
#if ASYNC_LDS
#if __has_builtin(__builtin_amdgcn_s_wait_asynccnt)
  __builtin_amdgcn_s_wait_asynccnt(0);
#else
  asm volatile("s_wait_asynccnt 0" ::: "memory");
#endif
#endif
}

// ---------------------------------------------------------------------------
// WMMA fragment loaders (16x16x32 bf16, contiguous k order used for both A and B).
// A (row-major M x K): lane<16 holds K{0..15}, lane>=16 K{16..31}.
// B from W^T (N x K row-major): lane<16 holds K{0..15}, lane>=16 K{16..31}.
// ---------------------------------------------------------------------------
__device__ __forceinline__ v16bf load_a_frag(const __bf16* A, int lda, int k0) {
  const int lane = threadIdx.x & 31;
  const __bf16* p = A + (long)(lane & 15) * lda + k0 + ((lane >> 4) << 4);
  v16bf v;
  float4* d = reinterpret_cast<float4*>(&v);
  d[0] = *reinterpret_cast<const float4*>(p);
  d[1] = *reinterpret_cast<const float4*>(p + 8);
  return v;
}
__device__ __forceinline__ v16bf load_b_frag(const __bf16* Bt, int ldb, int k0) {
  const int lane = threadIdx.x & 31;
  const __bf16* p = Bt + (long)(lane & 15) * ldb + k0 + ((lane >> 4) << 4);
  v16bf v;
  float4* d = reinterpret_cast<float4*>(&v);
  d[0] = *reinterpret_cast<const float4*>(p);
  d[1] = *reinterpret_cast<const float4*>(p + 8);
  return v;
}
// B fragment from an LDS tile with padded row stride (elements, mult of 8).
__device__ __forceinline__ v16bf load_b_frag_lds(const __bf16* bs, int nloc0,
                                                 int k0, int stride) {
  const int lane = threadIdx.x & 31;
  const __bf16* p = bs + (long)(nloc0 + (lane & 15)) * stride + k0 + ((lane >> 4) << 4);
  v16bf v;
  float4* d = reinterpret_cast<float4*>(&v);
  d[0] = *reinterpret_cast<const float4*>(p);
  d[1] = *reinterpret_cast<const float4*>(p + 8);
  return v;
}
__device__ __forceinline__ v8f wmma_bf16(v16bf a, v16bf b, v8f c) {
  return __builtin_amdgcn_wmma_f32_16x16x32_bf16(false, a, false, b, (short)0, c, false, false);
}
__device__ __forceinline__ float wave_sum(float v) {
#pragma unroll
  for (int m = 16; m >= 1; m >>= 1) v += __shfl_xor(v, m, 32);
  return v;
}
__device__ __forceinline__ float wave_max(float v) {
#pragma unroll
  for (int m = 16; m >= 1; m >>= 1) v = fmaxf(v, __shfl_xor(v, m, 32));
  return v;
}

// ---------------------------------------------------------------------------
// Weight prep: f32 (K x N, row stride src_ld) -> bf16 transposed (N x K).
// ---------------------------------------------------------------------------
__global__ void k_transpose_bf16(const float* __restrict__ src, __bf16* __restrict__ dst,
                                 int K, int N, int src_ld) {
  long tid = (long)blockIdx.x * blockDim.x + threadIdx.x;
  if (tid >= (long)K * N) return;
  int n = (int)(tid / K);
  int k = (int)(tid % K);
  dst[tid] = (__bf16)src[(long)k * src_ld + n];
}

// LayerNorm rows of (B*S, 256) -> bf16. One wave per row.
__global__ void k_layernorm(const float* __restrict__ obs, const float* __restrict__ g,
                            const float* __restrict__ bb, __bf16* __restrict__ out) {
  int wave = threadIdx.x >> 5, lane = threadIdx.x & 31;
  int row = blockIdx.x * 8 + wave;
  const float* x = obs + (long)row * OBS_;
  float s = 0.f, s2 = 0.f, vals[8];
#pragma unroll
  for (int j = 0; j < 8; ++j) {
    float v = x[lane + 32 * j];
    vals[j] = v; s += v; s2 += v * v;
  }
  s = wave_sum(s); s2 = wave_sum(s2);
  float mu = s * (1.f / 256.f);
  float var = s2 * (1.f / 256.f) - mu * mu;
  float rstd = rsqrtf(var + 1e-5f);
#pragma unroll
  for (int j = 0; j < 8; ++j) {
    int c = lane + 32 * j;
    out[(long)row * OBS_ + c] = (__bf16)((vals[j] - mu) * rstd * g[c] + bb[c]);
  }
}

// ---------------------------------------------------------------------------
// Generic GEMM: out = act(A[Mpad x K] @ Bt^T + bias). Block = 128 rows x 64
// cols (8 waves of 16 rows). B tile double-buffered in LDS via async copies.
// 8 fragments loaded + pinned live, then an 8-WMMA burst (single dscnt drain).
// ---------------------------------------------------------------------------
#define GEMM_BS_STRIDE 72   // 144B rows (mult of 16B), conflict-reduced
__global__ void k_gemm(const __bf16* __restrict__ A, int lda,
                       const __bf16* __restrict__ Bt, int ldb,
                       const float* __restrict__ bias,
                       __bf16* __restrict__ outb, float* __restrict__ outf, int ldo,
                       int Mdiv128, int Mvalid, int K, int relu) {
  __shared__ __bf16 Bs[2 * 64 * GEMM_BS_STRIDE];
  const int tid = threadIdx.x;
  const int wave = tid >> 5, lane = tid & 31;
  const int mb = blockIdx.x % Mdiv128;
  const int nb = blockIdx.x / Mdiv128;
  const __bf16* Arow = A + (long)(mb * 128 + wave * 16) * lda;
  const __bf16* Brow = Bt + (long)(nb * 64) * ldb;

  auto stage = [&](int kc, __bf16* bs) {
#pragma unroll
    for (int r = 0; r < 2; ++r) {
      int s = tid + 256 * r;              // 512 x 16B segments = 64n x 64k
      int kseg = s & 7, n = s >> 3;
      async_cp16(Brow + (long)n * ldb + kc + kseg * 8,
                 bs + n * GEMM_BS_STRIDE + kseg * 8);
    }
  };

  v8f acc[4];
#pragma unroll
  for (int t = 0; t < 4; ++t)
#pragma unroll
    for (int j = 0; j < 8; ++j) acc[t][j] = 0.f;

  const int nchunks = K >> 6;
  stage(0, Bs);
  wait_async_copies();
  __syncthreads();
  int buf = 0;
  for (int c = 0; c < nchunks; ++c) {
    __bf16* bs = Bs + buf * (64 * GEMM_BS_STRIDE);
    if (c + 1 < nchunks) stage((c + 1) << 6, Bs + (buf ^ 1) * (64 * GEMM_BS_STRIDE));
    int kc = c << 6;
    v16bf a0 = load_a_frag(Arow, lda, kc);
    v16bf a1 = load_a_frag(Arow, lda, kc + 32);
    v16bf b0 = load_b_frag_lds(bs,  0, 0,  GEMM_BS_STRIDE);
    v16bf b1 = load_b_frag_lds(bs, 16, 0,  GEMM_BS_STRIDE);
    v16bf b2 = load_b_frag_lds(bs, 32, 0,  GEMM_BS_STRIDE);
    v16bf b3 = load_b_frag_lds(bs, 48, 0,  GEMM_BS_STRIDE);
    v16bf b4 = load_b_frag_lds(bs,  0, 32, GEMM_BS_STRIDE);
    v16bf b5 = load_b_frag_lds(bs, 16, 32, GEMM_BS_STRIDE);
    v16bf b6 = load_b_frag_lds(bs, 32, 32, GEMM_BS_STRIDE);
    v16bf b7 = load_b_frag_lds(bs, 48, 32, GEMM_BS_STRIDE);
    pin2(a0, a1);
    pin8(b0, b1, b2, b3, b4, b5, b6, b7);   // all loads batch above this point
    acc[0] = wmma_bf16(a0, b0, acc[0]);
    acc[1] = wmma_bf16(a0, b1, acc[1]);
    acc[2] = wmma_bf16(a0, b2, acc[2]);
    acc[3] = wmma_bf16(a0, b3, acc[3]);
    acc[0] = wmma_bf16(a1, b4, acc[0]);
    acc[1] = wmma_bf16(a1, b5, acc[1]);
    acc[2] = wmma_bf16(a1, b6, acc[2]);
    acc[3] = wmma_bf16(a1, b7, acc[3]);
    wait_async_copies();
    __syncthreads();
    buf ^= 1;
  }
  int rbase = (lane >> 4) * 8, col = lane & 15;
#pragma unroll
  for (int t = 0; t < 4; ++t) {
    int n = nb * 64 + t * 16 + col;
    float bv = bias[n];
#pragma unroll
    for (int j = 0; j < 8; ++j) {
      int m = mb * 128 + wave * 16 + rbase + j;
      if (m < Mvalid) {
        float v = acc[t][j] + bv;
        if (relu) v = fmaxf(v, 0.f);
        if (outb) outb[(long)m * ldo + n] = (__bf16)v;
        else      outf[(long)m * ldo + n] = v;
      }
    }
  }
}

__global__ void k_zero_state(__bf16* hb, float* hf) {
  int i = blockIdx.x * blockDim.x + threadIdx.x;
  if (i < B_ * H_) { hb[i] = (__bf16)0.f; hf[i] = 0.f; }
}

// s0 = softmax((post_oe[:,0] + gumbel[0]) / tau). One wave per (b,l) group.
__global__ void k_init_s(const float* __restrict__ post_oe, const float* __restrict__ gum,
                         const int* __restrict__ tau_p, float* __restrict__ s_f,
                         __bf16* __restrict__ s_b) {
  int wave = threadIdx.x >> 5, lane = threadIdx.x & 31;
  int g = blockIdx.x * 8 + wave;     // 0..2047
  int b = g >> 5, l = g & 31;
  float tau = (float)tau_p[0];
  int idx = l * 32 + lane;
  float v = (post_oe[((long)b * S_) * STOCH_ + idx] + gum[(long)b * STOCH_ + idx]) / tau;
  float m = wave_max(v);
  float e = __expf(v - m);
  float p = e / wave_sum(e);
  s_f[b * STOCH_ + idx] = p;
  s_b[b * STOCH_ + idx] = (__bf16)p;
}

// ---------------------------------------------------------------------------
// One GRU scan step. Block owns 64 hidden columns; the 6 gate-weight tiles for
// a 32-wide K chunk are staged (async) into a double-buffered LDS tile shared
// by all 4 M-tiles. 6 fragments loaded + pinned per nt-half, 6-WMMA burst.
// Epilogue: one-hot action row gather + biases + GRU elementwise math.
// h buffers double-buffered across steps (cross-block WAR hazard).
// ---------------------------------------------------------------------------
#define GRU_BS_STRIDE 40    // 80B rows (mult of 16B)
__global__ void k_gru_step(int i, const int* __restrict__ act,
                           const float* __restrict__ gru_wi,   // f32, 1042 x 3072
                           const float* __restrict__ bi, const float* __restrict__ bh,
                           const __bf16* __restrict__ wist, const __bf16* __restrict__ wht,
                           const __bf16* __restrict__ s_b,
                           const __bf16* __restrict__ h_b_in, const float* __restrict__ h_f_in,
                           __bf16* __restrict__ h_b_out, float* __restrict__ h_f_out,
                           float* __restrict__ hs_out, __bf16* __restrict__ dec_in) {
  __shared__ __bf16 Bs[2 * 6 * 64 * GRU_BS_STRIDE];   // 2 x 30KB (24KB unpadded)
  const int tid = threadIdx.x;
  const int wave = tid >> 5, lane = tid & 31;
  const int m0 = (wave & 3) * 16;          // batch-row tile
  const int ch = wave >> 2;                // 32-col half
  const int ncolbase = blockIdx.x * 64;

  auto stage = [&](int kc, __bf16* bs) {
#pragma unroll
    for (int r = 0; r < 6; ++r) {
      int s = tid + 256 * r;               // 1536 x 16B = 6 gates x 64n x 32k
      int kseg = s & 3;
      int n = (s >> 2) & 63;
      int g2 = s >> 8;                     // 0..5
      const __bf16* src =
          (g2 < 3 ? wist + (long)(g2 * H_ + ncolbase + n) * H_
                  : wht  + (long)((g2 - 3) * H_ + ncolbase + n) * H_) + kc + kseg * 8;
      async_cp16(src, bs + ((g2 * 64 + n) * GRU_BS_STRIDE) + kseg * 8);
    }
  };

  v8f acc[6][2];
#pragma unroll
  for (int g2 = 0; g2 < 6; ++g2)
#pragma unroll
    for (int nt = 0; nt < 2; ++nt)
#pragma unroll
      for (int j = 0; j < 8; ++j) acc[g2][nt][j] = 0.f;

  stage(0, Bs);
  wait_async_copies();
  __syncthreads();
  int buf = 0;
  for (int c = 0; c < (H_ >> 5); ++c) {    // 32 chunks of K=32
    __bf16* bs = Bs + buf * (6 * 64 * GRU_BS_STRIDE);
    if (c + 1 < (H_ >> 5)) stage((c + 1) << 5, Bs + (buf ^ 1) * (6 * 64 * GRU_BS_STRIDE));
    int kc = c << 5;
    v16bf a_s = load_a_frag(s_b    + (long)m0 * H_, H_, kc);
    v16bf a_h = load_a_frag(h_b_in + (long)m0 * H_, H_, kc);
    pin2(a_s, a_h);
#pragma unroll
    for (int nt = 0; nt < 2; ++nt) {
      int nloc = ch * 32 + nt * 16;
      v16bf f0 = load_b_frag_lds(Bs + buf * (6*64*GRU_BS_STRIDE) + 0*(64*GRU_BS_STRIDE), nloc, 0, GRU_BS_STRIDE);
      v16bf f1 = load_b_frag_lds(Bs + buf * (6*64*GRU_BS_STRIDE) + 1*(64*GRU_BS_STRIDE), nloc, 0, GRU_BS_STRIDE);
      v16bf f2 = load_b_frag_lds(Bs + buf * (6*64*GRU_BS_STRIDE) + 2*(64*GRU_BS_STRIDE), nloc, 0, GRU_BS_STRIDE);
      v16bf f3 = load_b_frag_lds(Bs + buf * (6*64*GRU_BS_STRIDE) + 3*(64*GRU_BS_STRIDE), nloc, 0, GRU_BS_STRIDE);
      v16bf f4 = load_b_frag_lds(Bs + buf * (6*64*GRU_BS_STRIDE) + 4*(64*GRU_BS_STRIDE), nloc, 0, GRU_BS_STRIDE);
      v16bf f5 = load_b_frag_lds(Bs + buf * (6*64*GRU_BS_STRIDE) + 5*(64*GRU_BS_STRIDE), nloc, 0, GRU_BS_STRIDE);
      pin6(f0, f1, f2, f3, f4, f5);        // batch the 6 ds_load pairs
      acc[0][nt] = wmma_bf16(a_s, f0, acc[0][nt]);
      acc[1][nt] = wmma_bf16(a_s, f1, acc[1][nt]);
      acc[2][nt] = wmma_bf16(a_s, f2, acc[2][nt]);
      acc[3][nt] = wmma_bf16(a_h, f3, acc[3][nt]);
      acc[4][nt] = wmma_bf16(a_h, f4, acc[4][nt]);
      acc[5][nt] = wmma_bf16(a_h, f5, acc[5][nt]);
    }
    (void)bs;
    wait_async_copies();
    __syncthreads();
    buf ^= 1;
  }

  int rbase = (lane >> 4) * 8, col = lane & 15;
#pragma unroll
  for (int j = 0; j < 8; ++j) {
    int b = m0 + rbase + j;
    int a_idx = act[b * S_ + i];
    const float* wa = gru_wi + (long)(H_ + a_idx) * (3 * H_);  // one-hot row
#pragma unroll
    for (int nt = 0; nt < 2; ++nt) {
      int n = ncolbase + ch * 32 + nt * 16 + col;
      float ir = acc[0][nt][j] + bi[n]          + wa[n];
      float iz = acc[1][nt][j] + bi[H_ + n]     + wa[H_ + n];
      float inn= acc[2][nt][j] + bi[2 * H_ + n] + wa[2 * H_ + n];
      float hr = acc[3][nt][j] + bh[n];
      float hz = acc[4][nt][j] + bh[H_ + n];
      float hn = acc[5][nt][j] + bh[2 * H_ + n];
      float r  = 1.f / (1.f + __expf(-(ir + hr)));
      float z  = 1.f / (1.f + __expf(-(iz + hz)));
      float nn = tanhf(inn + r * hn);
      float hnew = (1.f - z) * nn + z * h_f_in[b * H_ + n];
      h_f_out[b * H_ + n] = hnew;
      h_b_out[b * H_ + n] = (__bf16)hnew;
      long rrow = (long)b * SM1 + i;
      hs_out[rrow * H_ + n]   = hnew;
      dec_in[rrow * 2048 + n] = (__bf16)hnew;    // hs half of decoder input
    }
  }
}

// ---------------------------------------------------------------------------
// Posterior step: logits = h_new @ post_w_top + post_oe[:, i+1] (C-init), then
// gumbel-softmax over C=32 in LDS. Depth-2 software pipeline on fragments.
// ---------------------------------------------------------------------------
__global__ void k_post_step(int i, const __bf16* __restrict__ h_b_new,
                            const float* __restrict__ post_oe,
                            const __bf16* __restrict__ postw_ht,
                            const float* __restrict__ gum, const int* __restrict__ tau_p,
                            float* __restrict__ post_logits,
                            float* __restrict__ s_f, __bf16* __restrict__ s_b,
                            __bf16* __restrict__ dec_in, float* __restrict__ samples_out) {
  __shared__ float smem[64][65];
  int wave = threadIdx.x >> 5, lane = threadIdx.x & 31;
  int m0 = (wave & 3) * 16;
  int np = wave >> 2;                  // 32-col half
  int nc = blockIdx.x;                 // 64-col chunk
  int rbase = (lane >> 4) * 8, col = lane & 15;
  const __bf16* Arow = h_b_new + (long)m0 * H_;
  const __bf16* B0 = postw_ht + (long)(nc * 64 + np * 32) * H_;
  const __bf16* B1 = B0 + 16 * H_;

  v8f acc[2];
#pragma unroll
  for (int nt = 0; nt < 2; ++nt) {
    int n = nc * 64 + np * 32 + nt * 16 + col;
#pragma unroll
    for (int j = 0; j < 8; ++j) {
      int b = m0 + rbase + j;
      acc[nt][j] = post_oe[((long)b * S_ + (i + 1)) * STOCH_ + n];  // C init
    }
  }
  v16bf a  = load_a_frag(Arow, H_, 0);
  v16bf b0 = load_b_frag(B0, H_, 0);
  v16bf b1 = load_b_frag(B1, H_, 0);
  for (int k0 = 0; k0 < H_; k0 += 32) {
    int kn = (k0 + 32 < H_) ? k0 + 32 : 0;      // prefetch next slice
    v16bf an  = load_a_frag(Arow, H_, kn);
    v16bf b0n = load_b_frag(B0, H_, kn);
    v16bf b1n = load_b_frag(B1, H_, kn);
    pin2(b0n, b1n);
    acc[0] = wmma_bf16(a, b0, acc[0]);
    acc[1] = wmma_bf16(a, b1, acc[1]);
    a = an; b0 = b0n; b1 = b1n;
  }
#pragma unroll
  for (int nt = 0; nt < 2; ++nt) {
    int nl = np * 32 + nt * 16 + col;
    int n  = nc * 64 + nl;
#pragma unroll
    for (int j = 0; j < 8; ++j) {
      int b = m0 + rbase + j;
      float v = acc[nt][j];
      post_logits[((long)b * SM1 + i) * STOCH_ + n] = v;
      smem[b][nl] = v;
    }
  }
  __syncthreads();

  float tau = (float)tau_p[0];
  for (int rep = 0; rep < 16; ++rep) {           // 128 (b,l) groups / 8 waves
    int g = wave * 16 + rep;
    int b = g >> 1, ll = g & 1;
    int l = nc * 2 + ll;
    int idx = l * 32 + lane;
    float v = (smem[b][ll * 32 + lane] +
               gum[((long)(i + 1) * B_ + b) * STOCH_ + idx]) / tau;
    float mx = wave_max(v);
    float e = __expf(v - mx);
    float p = e / wave_sum(e);
    float sold = s_f[b * STOCH_ + idx];
    long rrow = (long)b * SM1 + i;
    samples_out[rrow * STOCH_ + idx]   = sold;          // scan emits carry-in s
    dec_in[rrow * 2048 + STOCH_ + idx] = (__bf16)sold;  // samples half
    s_f[b * STOCH_ + idx] = p;
    s_b[b * STOCH_ + idx] = (__bf16)p;
  }
}

// KL per (b,t): one wave, lanes = C, loop over L. Fixed-order reductions.
__global__ void k_kl_pairs(const float* __restrict__ pl, const float* __restrict__ ql,
                           float* __restrict__ kl_bi) {
  int wid = blockIdx.x * 8 + (threadIdx.x >> 5);
  int lane = threadIdx.x & 31;
  long base = (long)wid * STOCH_;
  float fwd = 0.f, rev = 0.f;
  for (int l = 0; l < 32; ++l) {
    float xp = pl[base + l * 32 + lane];
    float xq = ql[base + l * 32 + lane];
    float mp = wave_max(xp);
    float lsep = mp + __logf(wave_sum(__expf(xp - mp)));
    float mq = wave_max(xq);
    float lseq = mq + __logf(wave_sum(__expf(xq - mq)));
    float lp = xp - lsep, lq = xq - lseq;
    float p = __expf(lp), q = __expf(lq);
    fwd += p * (lp - lq);
    rev += q * (lq - lp);
  }
  fwd = wave_sum(fwd); rev = wave_sum(rev);
  if (lane == 0) kl_bi[wid] = fmaxf(0.8f * fwd + 0.2f * rev - 1.0f, 0.f);
}
__global__ void k_kl_reduce(const float* __restrict__ kl_bi, float* __restrict__ out) {
  int wave = threadIdx.x >> 5, lane = threadIdx.x & 31;
  int b = blockIdx.x * 8 + wave;
  float s = 0.f;
  for (int t = lane; t < SM1; t += 32) s += kl_bi[b * SM1 + t];
  s = wave_sum(s);
  if (lane == 0) out[b] = s / (float)SM1;
}

// reward = dec_in @ rew_w + rew_b, read from f32 hs/samples outputs.
__global__ void k_reward(const float* __restrict__ hs, const float* __restrict__ samp,
                         const float* __restrict__ rw, const float* __restrict__ rb,
                         float* __restrict__ out) {
  int wid = blockIdx.x * 8 + (threadIdx.x >> 5);
  int lane = threadIdx.x & 31;
  long base = (long)wid * STOCH_;
  float acc = 0.f;
  for (int k = lane; k < STOCH_; k += 32)
    acc += hs[base + k] * rw[k] + samp[base + k] * rw[STOCH_ + k];
  acc = wave_sum(acc);
  if (lane == 0) out[wid] = acc + rb[0];
}

// ---------------------------------------------------------------------------
extern "C" void kernel_launch(void* const* d_in, const int* in_sizes, int n_in,
                              void* d_out, int out_size, void* d_ws, size_t ws_size,
                              hipStream_t stream) {
  (void)in_sizes; (void)n_in; (void)out_size; (void)ws_size;
  const float* obs    = (const float*)d_in[0];
  const int*   act    = (const int*)  d_in[1];
  const float* gum    = (const float*)d_in[2];
  const int*   tau    = (const int*)  d_in[3];
  const float* ln_g   = (const float*)d_in[4];
  const float* ln_b   = (const float*)d_in[5];
  const float* enc_w1 = (const float*)d_in[6];
  const float* enc_b1 = (const float*)d_in[7];
  const float* enc_w2 = (const float*)d_in[8];
  const float* enc_b2 = (const float*)d_in[9];
  const float* gru_wi = (const float*)d_in[10];
  const float* gru_wh = (const float*)d_in[11];
  const float* gru_bi = (const float*)d_in[12];
  const float* gru_bh = (const float*)d_in[13];
  const float* prior_w= (const float*)d_in[14];
  const float* prior_b= (const float*)d_in[15];
  const float* post_w = (const float*)d_in[16];
  const float* post_b = (const float*)d_in[17];
  const float* dec_w1 = (const float*)d_in[18];
  const float* dec_b1 = (const float*)d_in[19];
  const float* dec_w2 = (const float*)d_in[20];
  const float* dec_b2 = (const float*)d_in[21];
  const float* rew_w  = (const float*)d_in[22];
  const float* rew_b  = (const float*)d_in[23];

  float* out   = (float*)d_out;
  float* recon = out;                    // 16320 x 256
  float* rew   = out + 4177920;          // 16320
  float* klo   = out + 4194240;          // 64
  float* hs    = out + 4194304;          // 16320 x 1024
  float* samp  = out + 20905984;         // 16320 x 1024

  char* w = (char*)d_ws;
  size_t off = 0;
  auto alloc = [&](size_t bytes) -> char* {
    char* p = w + off; off += (bytes + 255) & ~(size_t)255; return p;
  };
  __bf16* enc_w1t = (__bf16*)alloc(1024ull * 256 * 2);
  __bf16* enc_w2t = (__bf16*)alloc(1024ull * 1024 * 2);
  __bf16* wist    = (__bf16*)alloc(3072ull * 1024 * 2);
  __bf16* wht     = (__bf16*)alloc(3072ull * 1024 * 2);
  __bf16* posthT  = (__bf16*)alloc(1024ull * 1024 * 2);
  __bf16* postoeT = (__bf16*)alloc(1024ull * 1024 * 2);
  __bf16* priorT  = (__bf16*)alloc(1024ull * 1024 * 2);
  __bf16* dec1t   = (__bf16*)alloc(1024ull * 2048 * 2);
  __bf16* dec2t   = (__bf16*)alloc(256ull  * 1024 * 2);
  __bf16* xln     = (__bf16*)alloc((size_t)MPAD * 256 * 2);
  char*   obsenc_r= alloc((size_t)MPAD * 1024 * 2);   // obs_enc bf16; reused as dec_hidden
  char*   postoe_r= alloc((size_t)MPAD * 1024 * 4);   // post_oe f32; reused as prior_logits
  char*   pl_r    = alloc((size_t)MPAD * 1024 * 4);   // enc_h1 bf16 pre-scan; post_logits f32
  __bf16* dec_in  = (__bf16*)alloc((size_t)MPAD * 2048 * 2);  // padded to 16384 rows
  __bf16* h_bf    = (__bf16*)alloc(2ull * B_ * H_ * 2);
  float*  h_f     = (float*) alloc(2ull * B_ * H_ * 4);
  __bf16* s_bf    = (__bf16*)alloc((size_t)B_ * H_ * 2);
  float*  s_f     = (float*) alloc((size_t)B_ * H_ * 4);
  float*  kl_bi   = (float*) alloc(16320ull * 4);

  __bf16* obs_enc    = (__bf16*)obsenc_r;
  __bf16* dec_hidden = (__bf16*)obsenc_r;
  float*  post_oe    = (float*)postoe_r;
  float*  prior_log  = (float*)postoe_r;
  __bf16* enc_h1     = (__bf16*)pl_r;
  float*  post_log   = (float*)pl_r;

  // --- Weight prep (transpose + bf16) ---
  k_transpose_bf16<<<(256*1024)/256,  256, 0, stream>>>(enc_w1, enc_w1t, 256, 1024, 1024);
  k_transpose_bf16<<<(1024*1024)/256, 256, 0, stream>>>(enc_w2, enc_w2t, 1024, 1024, 1024);
  k_transpose_bf16<<<(1024*3072)/256, 256, 0, stream>>>(gru_wi, wist, 1024, 3072, 3072);
  k_transpose_bf16<<<(1024*3072)/256, 256, 0, stream>>>(gru_wh, wht, 1024, 3072, 3072);
  k_transpose_bf16<<<(1024*1024)/256, 256, 0, stream>>>(post_w, posthT, 1024, 1024, 1024);
  k_transpose_bf16<<<(1024*1024)/256, 256, 0, stream>>>(post_w + 1024*1024, postoeT, 1024, 1024, 1024);
  k_transpose_bf16<<<(1024*1024)/256, 256, 0, stream>>>(prior_w, priorT, 1024, 1024, 1024);
  k_transpose_bf16<<<(2048*1024)/256, 256, 0, stream>>>(dec_w1, dec1t, 2048, 1024, 1024);
  k_transpose_bf16<<<(1024*256)/256,  256, 0, stream>>>(dec_w2, dec2t, 1024, 256, 256);

  // --- LayerNorm + encoder + batched posterior obs_enc contribution ---
  k_layernorm<<<2048, 256, 0, stream>>>(obs, ln_g, ln_b, xln);
  k_gemm<<<128*16, 256, 0, stream>>>(xln, 256,  enc_w1t, 256,  enc_b1, enc_h1, nullptr, 1024, 128, 16384, 256,  1);
  k_gemm<<<128*16, 256, 0, stream>>>(enc_h1, 1024, enc_w2t, 1024, enc_b2, obs_enc, nullptr, 1024, 128, 16384, 1024, 1);
  k_gemm<<<128*16, 256, 0, stream>>>(obs_enc, 1024, postoeT, 1024, post_b, nullptr, post_oe, 1024, 128, 16384, 1024, 0);

  // --- Init state ---
  k_zero_state<<<256, 256, 0, stream>>>(h_bf, h_f);          // slot 0
  k_init_s<<<256, 256, 0, stream>>>(post_oe, gum, tau, s_f, s_bf);

  // --- Sequential GRU scan (255 steps, 2 fused kernels each) ---
  for (int i = 0; i < SM1; ++i) {
    int pi = i & 1, po = (i + 1) & 1;
    k_gru_step<<<16, 256, 0, stream>>>(i, act, gru_wi, gru_bi, gru_bh, wist, wht,
                                       s_bf, h_bf + pi * (B_*H_), h_f + pi * (B_*H_),
                                       h_bf + po * (B_*H_), h_f + po * (B_*H_),
                                       hs, dec_in);
    k_post_step<<<16, 256, 0, stream>>>(i, h_bf + po * (B_*H_), post_oe, posthT,
                                        gum, tau, post_log, s_f, s_bf, dec_in, samp);
  }

  // --- Prior, KL, decoder, reward ---
  k_gemm<<<128*16, 256, 0, stream>>>(dec_in, 2048, priorT, 1024, prior_b, nullptr, prior_log, 1024, 128, 16320, 1024, 0);
  k_kl_pairs<<<2040, 256, 0, stream>>>(post_log, prior_log, kl_bi);
  k_kl_reduce<<<8, 256, 0, stream>>>(kl_bi, klo);
  k_gemm<<<128*16, 256, 0, stream>>>(dec_in, 2048, dec1t, 2048, dec_b1, dec_hidden, nullptr, 1024, 128, 16320, 2048, 1);
  k_gemm<<<128*4,  256, 0, stream>>>(dec_hidden, 1024, dec2t, 1024, dec_b2, nullptr, recon, 256, 128, 16320, 1024, 0);
  k_reward<<<2040, 256, 0, stream>>>(hs, samp, rew_w, rew_b, rew);
}
